// MultiAttention_80229989089874
// MI455X (gfx1250) — matrix operand order repaired and checked by reading, not compiled, not run on hardware
//
#include <hip/hip_runtime.h>

// ---------------------------------------------------------------------------
// Multi-head attention forward on gfx1250 (MI455X), f16 WMMA w/ f32 accum.
// ~137 GFLOP vs ~150MB HBM -> compute-bound; drive v_wmma_f32_16x16x32_f16.
// K/V tiles staged block-wide via the Tensor Data Mover (TENSORcnt),
// double-buffered in LDS; softmax scale folded into the Q projection.
// ---------------------------------------------------------------------------

typedef __attribute__((ext_vector_type(16))) _Float16 v16h;
typedef __attribute__((ext_vector_type(8)))  _Float16 v8h;
typedef __attribute__((ext_vector_type(4)))  _Float16 v4h;
typedef __attribute__((ext_vector_type(8)))  float    v8f;
typedef __attribute__((ext_vector_type(4)))  float    v4f;
typedef __attribute__((ext_vector_type(4)))  unsigned int u32x4;
typedef __attribute__((ext_vector_type(4)))  int      i32x4;
typedef __attribute__((ext_vector_type(8)))  int      i32x8;

#define EMB  1024
#define SEQ  2048
#define NB   4
#define NH   16
#define DH   64
#define MTOT (NB * SEQ)   // 8192 rows

// Build one lane's 16-half fragment from two 16-byte chunks.
// A-frag (16x32 f16, row = lane&15): p0 = rowptr + half*16, p1 = p0 + 8
// B-frag (32x16 f16, col = lane&15, source = B^T row-major [N,K]):
//   p0 = rowptr + half*16, p1 = p0 + 8
__device__ __forceinline__ v16h ld_frag(const _Float16* p0, const _Float16* p1) {
  v8h a = *(const v8h*)p0;
  v8h b = *(const v8h*)p1;
  return __builtin_shufflevector(a, b, 0,1,2,3,4,5,6,7,8,9,10,11,12,13,14,15);
}

__device__ __forceinline__ v8f wmma_f16(v16h a, v16h b, v8f c) {
  return __builtin_amdgcn_wmma_f32_16x16x32_f16(false, a, false, b, (short)0, c,
                                                false, false);
}

// XOR-lane shuffles via ds_swizzle (group-of-32: imm = xor<<10 | or<<5 | and).
template <int IMM>
__device__ __forceinline__ float swz(float x) {
  int i = __builtin_amdgcn_ds_swizzle(__builtin_bit_cast(int, x), IMM);
  return __builtin_bit_cast(float, i);
}
// Reduce across the 16 lanes of each half-wave (masks 1,2,4,8 stay in-half).
__device__ __forceinline__ float rmax16(float x) {
  x = fmaxf(x, swz<0x041F>(x));
  x = fmaxf(x, swz<0x081F>(x));
  x = fmaxf(x, swz<0x101F>(x));
  x = fmaxf(x, swz<0x201F>(x));
  return x;
}
__device__ __forceinline__ float rsum16(float x) {
  x += swz<0x041F>(x);
  x += swz<0x081F>(x);
  x += swz<0x101F>(x);
  x += swz<0x201F>(x);
  return x;
}

// ---------------------------------------------------------------------------
// Tensor Data Mover: 2D tile (f16) global -> LDS.  D# per CDNA5 ISA ch.8.
// Group0: [1:0]=count=1, [63:32]=lds_addr, [120:64]=global_addr, [127:126]=2.
// Group1: data_size=1 (2B) @ [17:16]; tensor_dim0 @ [79:48]; tensor_dim1 @
// [111:80]; tile_dim0 @ [127:112]; tile_dim1 @ [143:128]; stride0 @ [207:160].
// Dims set huge so the OOB clip never triggers; groups 2/3 = 0 (2-D tile).
// This toolchain's builtin is the 6-operand form (extra i32x8, then cpol).
// ---------------------------------------------------------------------------
__device__ __forceinline__ void tdm_load_2d_f16(unsigned lds_off,
                                                const void* gptr,
                                                unsigned tile0, unsigned tile1,
                                                unsigned stride0_elems) {
  unsigned long long ga = (unsigned long long)gptr;
  u32x4 g0;
  g0[0] = 1u;                                                  // count = 1
  g0[1] = lds_off;                                             // lds_addr
  g0[2] = (unsigned)ga;                                        // gaddr[31:0]
  g0[3] = (unsigned)((ga >> 32) & 0x01FFFFFFu) | (2u << 30);   // gaddr[56:32],type=2
  const unsigned dim0 = 1u << 24, dim1 = 1u << 24;             // never clip
  i32x8 g1;
  g1[0] = (int)(1u << 16);                                     // data_size = 2B
  g1[1] = (int)((dim0 & 0xFFFFu) << 16);                       // dim0 lo16
  g1[2] = (int)((dim0 >> 16) | ((dim1 & 0xFFFFu) << 16));      // dim0 hi | dim1 lo
  g1[3] = (int)((dim1 >> 16) | (tile0 << 16));                 // dim1 hi | tile0
  g1[4] = (int)(tile1 & 0xFFFFu);                              // tile1, tile2=0
  g1[5] = (int)stride0_elems;                                  // stride0 lo32
  g1[6] = 0;                                                   // stride0 hi16
  g1[7] = 0;
  i32x4 z4 = {0, 0, 0, 0};
  i32x8 z8 = {0, 0, 0, 0, 0, 0, 0, 0};
  __builtin_amdgcn_tensor_load_to_lds(g0, g1, z4, z4, z8, 0);
}

// ---------------------------------------------------------------------------
// Elementwise f32 -> f16 (4-wide)
// ---------------------------------------------------------------------------
__global__ void f32_to_f16_v4(const float* __restrict__ in,
                              _Float16* __restrict__ out, int n4) {
  int i = blockIdx.x * blockDim.x + threadIdx.x;
  if (i >= n4) return;
  v4f f = ((const v4f*)in)[i];
  v4h h = { (_Float16)f.x, (_Float16)f.y, (_Float16)f.z, (_Float16)f.w };
  ((v4h*)out)[i] = h;
}

// ---------------------------------------------------------------------------
// GEMM: C[m,n] = (sum_k A[m,k]*Bt[n,k] + bias[n]) * scale
//   A  : f16 [MTOT, EMB] row-major
//   Bt : f16 [EMB, EMB] row-major ([N,K] == torch Linear weight == WMMA B^T)
// MODE 0: C -> f16 row-major [MTOT, EMB]            (Q [scale=1/8], K)
// MODE 1: C -> f16 transposed per head [B,H,DH,SEQ] (V)
// MODE 2: C -> f32 row-major [MTOT, EMB]            (final output)
// Block: 256 thr = 8 waves as 4(M) x 2(N); wave tile 32x32 (2x2 C-frags).
// ---------------------------------------------------------------------------
template <int MODE>
__global__ __launch_bounds__(256) void gemm_wmma(
    const _Float16* __restrict__ A, const _Float16* __restrict__ Bt,
    const float* __restrict__ bias, float scale, void* __restrict__ Cout) {
  const int lane = threadIdx.x & 31;
  const int wid  = threadIdx.x >> 5;
  const int half = lane >> 4;
  const int r    = lane & 15;
  const int m0 = blockIdx.x * 128 + (wid & 3) * 32;
  const int n0 = blockIdx.y * 64  + (wid >> 2) * 32;

  v8f acc[2][2] = {};
  const _Float16* a0p = A  + (size_t)(m0 +      r) * EMB + half * 16;
  const _Float16* a1p = A  + (size_t)(m0 + 16 + r) * EMB + half * 16;
  const _Float16* b0p = Bt + (size_t)(n0 +      r) * EMB + half * 16;
  const _Float16* b1p = Bt + (size_t)(n0 + 16 + r) * EMB + half * 16;

  for (int kk = 0; kk < EMB; kk += 32) {
    v16h fa0 = ld_frag(a0p + kk, a0p + kk + 8);
    v16h fa1 = ld_frag(a1p + kk, a1p + kk + 8);
    v16h fb0 = ld_frag(b0p + kk, b0p + kk + 8);
    v16h fb1 = ld_frag(b1p + kk, b1p + kk + 8);
    acc[0][0] = wmma_f16(fa0, fb0, acc[0][0]);
    acc[0][1] = wmma_f16(fa0, fb1, acc[0][1]);
    acc[1][0] = wmma_f16(fa1, fb0, acc[1][0]);
    acc[1][1] = wmma_f16(fa1, fb1, acc[1][1]);
  }

  // C-frag layout: VGPR v, lanes 0-15 -> row v, lanes 16-31 -> row v+8; col = r.
  #pragma unroll
  for (int j = 0; j < 2; ++j) {
    const int col = n0 + j * 16 + r;
    const float bj = bias[col];
    #pragma unroll
    for (int i = 0; i < 2; ++i) {
      const int rowbase = m0 + i * 16 + half * 8;
      #pragma unroll
      for (int v = 0; v < 8; ++v) {
        const int row = rowbase + v;
        const float val = (acc[i][j][v] + bj) * scale;
        if (MODE == 0) {
          ((_Float16*)Cout)[(size_t)row * EMB + col] = (_Float16)val;
        } else if (MODE == 1) {
          const int b = row >> 11, s = row & (SEQ - 1);
          const int h = col >> 6,  d = col & (DH - 1);
          ((_Float16*)Cout)[((size_t)(b * NH + h) * DH + d) * SEQ + s] =
              (_Float16)val;
        } else {
          ((float*)Cout)[(size_t)row * EMB + col] = val;
        }
      }
    }
  }
}

// ---------------------------------------------------------------------------
// Flash attention. One block = 8 waves = 128 consecutive q rows of one (b,h);
// each wave owns 16 q rows.  Per 32-key window the block's K (32x64) and
// V^T (64x32) tiles are TDM-loaded into double-buffered LDS by wave 0
// (TENSORcnt + s_barrier), then all waves run:
//   S(16x32) = Qs(16x64)·K^T  ->  online softmax  ->  O(16x64) += P·V.
// Q was pre-scaled by 1/sqrt(DH) in its projection epilogue.
// ---------------------------------------------------------------------------
__global__ __launch_bounds__(256) void flash_attn(
    const _Float16* __restrict__ Qh, const _Float16* __restrict__ Kh,
    const _Float16* __restrict__ Vt, _Float16* __restrict__ Oh) {
  __shared__ _Float16 ldsK[2][32 * DH];   // [buf][s_rel*64 + d]      8 KB
  __shared__ _Float16 ldsV[2][DH * 32];   // [buf][d*32 + s_rel]      8 KB
  __shared__ _Float16 ldsP[8][16 * 32];   // per-wave P tile          8 KB

  const int lane = threadIdx.x & 31;
  const int wid  = threadIdx.x >> 5;
  const int half = lane >> 4;
  const int r    = lane & 15;
  const int bh   = (int)blockIdx.x >> 4;          // b*NH + h (16 blocks per bh)
  const int q0   = (((int)blockIdx.x & 15) * 8 + wid) * 16;
  const int b    = bh >> 4;
  const int h    = bh & 15;
  _Float16* P = ldsP[wid];

  const _Float16* kgbase = Kh + (size_t)(b * SEQ) * EMB + h * DH;
  const _Float16* vgbase = Vt + (size_t)bh * DH * SEQ;

  // Q fragments (A, K-dim = d): two 16x32 frags covering d=0..63.
  const _Float16* qp = Qh + (size_t)(b * SEQ + q0 + r) * EMB + h * DH + half * 16;
  const v16h aq0 = ld_frag(qp,      qp + 8);
  const v16h aq1 = ld_frag(qp + 32, qp + 40);

  v8f o[4] = {};
  float m_i[8], l_i[8];
  #pragma unroll
  for (int v = 0; v < 8; ++v) { m_i[v] = -1e30f; l_i[v] = 0.f; }

  // TDM issue: K window (32 rows x 64 cols, row stride EMB) and
  //            V window (64 rows x 32 cols, row stride SEQ).
  auto issue = [&](int kc, int buf) {
    tdm_load_2d_f16((unsigned)(size_t)&ldsK[buf][0], kgbase + (size_t)kc * EMB,
                    /*tile0=*/DH, /*tile1=*/32, /*stride0=*/EMB);
    tdm_load_2d_f16((unsigned)(size_t)&ldsV[buf][0], vgbase + kc,
                    /*tile0=*/32, /*tile1=*/DH, /*stride0=*/SEQ);
  };
  if (wid == 0) issue(0, 0);

  for (int kc = 0; kc < SEQ; kc += 32) {
    const int buf = (kc >> 5) & 1;
    if (wid == 0) {
      if (kc + 32 < SEQ) {
        issue(kc + 32, buf ^ 1);
        __builtin_amdgcn_s_wait_tensorcnt(2);  // current window's 2 TDMs done
      } else {
        __builtin_amdgcn_s_wait_tensorcnt(0);
      }
    }
    __syncthreads();  // publish LDS tiles to all waves

    // ---- S tile 16x32: two N=16 column tiles, each two K=32 WMMAs over d ----
    v8f s0 = {}, s1 = {};
    {
      const _Float16* kp0 = &ldsK[buf][0] + (size_t)r * DH + half * 16;
      const _Float16* kp1 = kp0 + 16 * DH;
      v16h bk00 = ld_frag(kp0,      kp0 + 8);
      v16h bk01 = ld_frag(kp0 + 32, kp0 + 40);
      v16h bk10 = ld_frag(kp1,      kp1 + 8);
      v16h bk11 = ld_frag(kp1 + 32, kp1 + 40);
      s0 = wmma_f16(aq0, bk00, s0);
      s0 = wmma_f16(aq1, bk01, s0);
      s1 = wmma_f16(aq0, bk10, s1);
      s1 = wmma_f16(aq1, bk11, s1);
    }

    // ---- online softmax (row v+half*8 lives in the lane's own half) ----
    float alpha[8];
    #pragma unroll
    for (int v = 0; v < 8; ++v) {
      const float x0 = s0[v];
      const float x1 = s1[v];
      const float mx = rmax16(fmaxf(x0, x1));
      const float mn = fmaxf(m_i[v], mx);
      const float p0 = __expf(x0 - mn);
      const float p1 = __expf(x1 - mn);
      const float rs = rsum16(p0 + p1);
      alpha[v] = __expf(m_i[v] - mn);
      l_i[v] = l_i[v] * alpha[v] + rs;
      m_i[v] = mn;
      P[(v + half * 8) * 32 + r]      = (_Float16)p0;  // C-layout -> LDS
      P[(v + half * 8) * 32 + 16 + r] = (_Float16)p1;
    }
    #pragma unroll
    for (int f = 0; f < 4; ++f)
      #pragma unroll
      for (int v = 0; v < 8; ++v) o[f][v] *= alpha[v];

    // Re-fragment P as WMMA A (16x32). Same-wave LDS is in-order; wait DScnt.
    asm volatile("s_wait_dscnt 0" ::: "memory");
    const _Float16* pp = P + r * 32 + half * 16;
    v16h ap = ld_frag(pp, pp + 8);

    // ---- O(16x64) += P(16x32) · V(32x64): B-frags from LDS V^T (d, s) ----
    #pragma unroll
    for (int f = 0; f < 4; ++f) {
      const _Float16* vp = &ldsV[buf][0] + (size_t)(f * 16 + r) * 32 + half * 16;
      v16h bv = ld_frag(vp, vp + 8);
      o[f] = wmma_f16(ap, bv, o[f]);
    }
    __syncthreads();  // all waves done with buf before TDM overwrites it
  }

  // ---- normalize and store f16 [B,S,EMB] ----
  _Float16* op = Oh + (size_t)(b * SEQ + q0 + half * 8) * EMB + h * DH;
  #pragma unroll
  for (int v = 0; v < 8; ++v) {
    const float inv = 1.f / l_i[v];
    #pragma unroll
    for (int f = 0; f < 4; ++f)
      op[(size_t)v * EMB + f * 16 + r] = (_Float16)(o[f][v] * inv);
  }
}

// ---------------------------------------------------------------------------
extern "C" void kernel_launch(void* const* d_in, const int* in_sizes, int n_in,
                              void* d_out, int out_size, void* d_ws,
                              size_t ws_size, hipStream_t stream) {
  const float* X  = (const float*)d_in[0];
  const float* Wq = (const float*)d_in[1];
  const float* bq = (const float*)d_in[2];
  const float* Wk = (const float*)d_in[3];
  const float* bk = (const float*)d_in[4];
  const float* Wv = (const float*)d_in[5];
  const float* bv = (const float*)d_in[6];
  const float* Wo = (const float*)d_in[7];
  const float* bo = (const float*)d_in[8];

  char* ws = (char*)d_ws;
  size_t off = 0;
  auto alloc = [&](size_t bytes) {
    char* p = ws + off;
    off += (bytes + 255) & ~(size_t)255;
    return p;
  };
  _Float16* Xh  = (_Float16*)alloc((size_t)MTOT * EMB * 2);
  _Float16* Wqh = (_Float16*)alloc((size_t)EMB * EMB * 2);
  _Float16* Wkh = (_Float16*)alloc((size_t)EMB * EMB * 2);
  _Float16* Wvh = (_Float16*)alloc((size_t)EMB * EMB * 2);
  _Float16* Woh = (_Float16*)alloc((size_t)EMB * EMB * 2);
  _Float16* Qh  = (_Float16*)alloc((size_t)MTOT * EMB * 2);
  _Float16* Kh  = (_Float16*)alloc((size_t)MTOT * EMB * 2);
  _Float16* Vt  = (_Float16*)alloc((size_t)MTOT * EMB * 2);
  _Float16* Ohb = (_Float16*)alloc((size_t)MTOT * EMB * 2);

  const int thr = 256;
  auto cvt = [&](const float* src, _Float16* dst, int n) {
    int n4 = n / 4;
    f32_to_f16_v4<<<(n4 + thr - 1) / thr, thr, 0, stream>>>(src, dst, n4);
  };
  cvt(X,  Xh,  MTOT * EMB);
  cvt(Wq, Wqh, EMB * EMB);
  cvt(Wk, Wkh, EMB * EMB);
  cvt(Wv, Wvh, EMB * EMB);
  cvt(Wo, Woh, EMB * EMB);

  dim3 gg(MTOT / 128, EMB / 64);
  const float iscale = 0.125f;  // 1/sqrt(DH), folded into Q projection
  gemm_wmma<0><<<gg, thr, 0, stream>>>(Xh, Wqh, bq, iscale, Qh);
  gemm_wmma<0><<<gg, thr, 0, stream>>>(Xh, Wkh, bk, 1.0f,   Kh);
  gemm_wmma<1><<<gg, thr, 0, stream>>>(Xh, Wvh, bv, 1.0f,   Vt);

  flash_attn<<<NB * NH * (SEQ / 128), thr, 0, stream>>>(Qh, Kh, Vt, Ohb);

  gemm_wmma<2><<<gg, thr, 0, stream>>>(Ohb, Woh, bo, 1.0f, d_out);
}
